// Decoder_84696755077495
// MI455X (gfx1250) — matrix operand order repaired and checked by reading, not compiled, not run on hardware
//
#include <hip/hip_runtime.h>

// ---------------- problem constants (from reference) ----------------
#define BB   2
#define NN   2048
#define KK   32
#define FV   148    // 3*16 + 100 node features
#define SSC  128    // h_S scalar dim
#define FE   35     // 3*1 + 32 edge features

typedef __attribute__((ext_vector_type(16))) __bf16 v16bf;
typedef __attribute__((ext_vector_type(8)))  float  v8f;

union ABFrag { unsigned short u[16]; v16bf v; };

__device__ __forceinline__ unsigned short f2bf(float f) {
  union { float f; unsigned u; } x; x.f = f;
  unsigned r = x.u + 0x7FFFu + ((x.u >> 16) & 1u);
  return (unsigned short)(r >> 16);
}

__device__ __forceinline__ float sigmoidf_(float x) { return 1.f / (1.f + __expf(-x)); }

// A-fragment (16x32 bf16) from an LDS row-major tile [16*? rows][ldk halves].
// lane<16 : row M=lane,    K = kt*32 + {0..15}
// lane>=16: row M=lane-16, K = kt*32 + {16..31}
__device__ __forceinline__ v16bf load_afrag(const unsigned short* tile, int ldk,
                                            int mtile, int kt, int lane) {
  int m  = mtile * 16 + (lane & 15);
  int kb = kt * 32 + ((lane >> 4) & 1) * 16;
  const unsigned short* p = tile + m * ldk + kb;
  ABFrag r;
#pragma unroll
  for (int j = 0; j < 8; ++j) r.u[j] = p[j];
#pragma unroll
  for (int j = 0; j < 8; ++j) r.u[8 + j] = p[8 + j];
  return r.v;
}

// B-fragment (32x16 bf16) pre-packed: frag = 32 lanes x 16 contiguous halves.
__device__ __forceinline__ v16bf load_bfrag(const unsigned short* packed, int frag, int lane) {
  const unsigned short* p = packed + (size_t)frag * 512 + lane * 16;
  ABFrag r;
#pragma unroll
  for (int j = 0; j < 16; ++j) r.u[j] = p[j];
  return r.v;
}

// Pack ws (Kreal x Nreal, row-major f32) into WMMA B-fragment layout (bf16).
// lane l holds ws[kt*32 + (l>=16)*16 + j][nt*16 + (l&15)], j=0..15.
__global__ __launch_bounds__(512) void pack_kernel(const float* __restrict__ w,
                                                   int Kreal, int Nreal, int KT,
                                                   unsigned short* __restrict__ dst) {
  int frag = blockIdx.x;
  int kt = frag % KT;
  int nt = frag / KT;
  int t  = threadIdx.x;
  int lp = t >> 4, j = t & 15;
  int k  = kt * 32 + ((lp >= 16) ? 16 : 0) + j;
  int nn = nt * 16 + (lp & 15);
  float v = (k < Kreal && nn < Nreal) ? w[(size_t)k * Nreal + nn] : 0.f;
  dst[(size_t)frag * 512 + t] = f2bf(v);
}

// ---------------- per-node edge message kernel ----------------
struct EdgeArgs {
  const float* hV; const float* hV0; const float* hS; const float* hE;
  const int* Eidx; const float* mask;
  const float *wh1, *wv1, *bs1;
  const float *wh2, *wv2, *bs2;
  const float *wh3, *wv3, *bs3;
  const unsigned short *pE1, *pE2, *pE3;
  float* hMid;
};

__global__ __launch_bounds__(64) void edge_kernel(EdgeArgs A) {
  __shared__ unsigned short tA1[32 * 416];  // GVP1 A-tile (ld 416); reused as GVP3 A-tile (ld 128)
  __shared__ unsigned short tA2[32 * 128];  // GVP2 A-tile
  __shared__ float vA[32 * 99];             // m-vectors -> vmu1 -> vmu2 -> vmu3
  __shared__ float vB[32 * 99];             // vh1 -> vh2 -> vh3
  __shared__ float acc_s[112];
  __shared__ float acc_v[48];
  __shared__ float red[4];

  const int tid  = threadIdx.x;
  const int lane = tid & 31;
  const int wid  = tid >> 5;   // M-tile (edges 0-15 / 16-31)
  const int nf   = blockIdx.x; // flat node
  const int b    = nf / NN;
  const int n    = nf % NN;
  const int e    = tid & 31;   // edge this thread helps with
  const int rep  = tid >> 5;   // 0/1 split of per-edge feature work

  for (int i = tid; i < 112; i += 64) acc_s[i] = 0.f;
  for (int i = tid; i < 48;  i += 64) acc_v[i] = 0.f;

  // ---- P1: gather + mask-blend -> m (scalars bf16 in tA1, vectors f32 in vA) ----
  {
    const int   j   = A.Eidx[(size_t)nf * KK + e];
    const float a   = A.mask[nf];
    const float att = (j < n) ? 1.f : 0.f;
    const float bw  = a * att, fw = a * (1.f - att);
    const float* hVn  = A.hV  + (size_t)nf * FV;
    const float* hVj  = A.hV  + ((size_t)b * NN + j) * FV;
    const float* hV0j = A.hV0 + ((size_t)b * NN + j) * FV;
    const float* hSj  = A.hS  + ((size_t)b * NN + j) * SSC;
    const float* hEe  = A.hE  + ((size_t)nf * KK + e) * FE;
    unsigned short* row = tA1 + e * 416;
    for (int f = rep; f < 100; f += 2) row[f]       = f2bf(hVn[48 + f]);
    for (int f = rep; f < 32;  f += 2) row[100 + f] = f2bf(a * hEe[3 + f]);
    for (int f = rep; f < 128; f += 2) row[132 + f] = f2bf(bw * hSj[f]);
    for (int f = rep; f < 100; f += 2) row[260 + f] = f2bf(bw * hVj[48 + f] + fw * hV0j[48 + f]);
    for (int f = rep; f < 23;  f += 2) row[393 + f] = 0;           // K-pad 393..415
    for (int f = rep; f < 28;  f += 2) tA2[e * 128 + 100 + f] = 0; // tA2 pad (vn2 overwrites 100..115)
    float* mv = vA + e * 99;
    for (int i = rep; i < 16; i += 2)
      for (int c = 0; c < 3; ++c) mv[i * 3 + c] = hVn[i * 3 + c];
    if (rep == 0)
      for (int c = 0; c < 3; ++c) mv[48 + c] = a * hEe[c];
    for (int i = rep; i < 16; i += 2)
      for (int c = 0; c < 3; ++c)
        mv[(17 + i) * 3 + c] = bw * hVj[i * 3 + c] + fw * hV0j[i * 3 + c];
  }
  __syncthreads();

  // ---- P2: GVP1 vh (33x33), vn -> tA1[360..392] ----
  for (int h = rep; h < 33; h += 2) {
    float x = 0, y = 0, z = 0;
    const float* mv = vA + e * 99;
    for (int i = 0; i < 33; ++i) {
      float w = A.wh1[i * 33 + h];
      x += mv[i * 3 + 0] * w; y += mv[i * 3 + 1] * w; z += mv[i * 3 + 2] * w;
    }
    float* o = vB + e * 99 + h * 3;
    o[0] = x; o[1] = y; o[2] = z;
    tA1[e * 416 + 360 + h] = f2bf(sqrtf(x * x + y * y + z * z + 1e-8f));
  }
  __syncthreads();

  // ---- P3: GVP1 vmu (gated) -> vA ----
  for (int o = rep; o < 16; o += 2) {
    float x = 0, y = 0, z = 0;
    const float* vh = vB + e * 99;
    for (int h = 0; h < 33; ++h) {
      float w = A.wv1[h * 16 + o];
      x += vh[h * 3 + 0] * w; y += vh[h * 3 + 1] * w; z += vh[h * 3 + 2] * w;
    }
    float g = sigmoidf_(sqrtf(x * x + y * y + z * z + 1e-8f));
    float* d = vA + e * 99 + o * 3;
    d[0] = x * g; d[1] = y * g; d[2] = z * g;
  }
  __syncthreads();

  // ---- P4: GVP1 scalar matmul (WMMA bf16, K=416) -> relu -> tA2 ----
  for (int nt = 0; nt < 7; ++nt) {
    v8f acc = {};
    for (int kt = 0; kt < 13; ++kt) {
      v16bf am = load_afrag(tA1, 416, wid, kt, lane);
      v16bf bm = load_bfrag(A.pE1, nt * 13 + kt, lane);
      acc = __builtin_amdgcn_wmma_f32_16x16x32_bf16(false, am, false, bm, (short)0, acc, false, false);
    }
    int fcol  = nt * 16 + (lane & 15);
    int ebase = wid * 16 + ((lane >> 4) & 1) * 8;
    if (fcol < 100) {
      float bias = A.bs1[fcol];
#pragma unroll
      for (int r = 0; r < 8; ++r)
        tA2[(ebase + r) * 128 + fcol] = f2bf(fmaxf(acc[r] + bias, 0.f));
    }
  }
  __syncthreads();

  // ---- P5: GVP2 vh/vn -> tA2[100..115]; zero GVP3 tile pads (tA1 reused, ld 128) ----
  for (int f = rep; f < 28; f += 2) tA1[e * 128 + 100 + f] = 0;
  for (int h = rep; h < 16; h += 2) {
    float x = 0, y = 0, z = 0;
    const float* mv = vA + e * 99;
    for (int i = 0; i < 16; ++i) {
      float w = A.wh2[i * 16 + h];
      x += mv[i * 3 + 0] * w; y += mv[i * 3 + 1] * w; z += mv[i * 3 + 2] * w;
    }
    float* o = vB + e * 99 + h * 3;
    o[0] = x; o[1] = y; o[2] = z;
    tA2[e * 128 + 100 + h] = f2bf(sqrtf(x * x + y * y + z * z + 1e-8f));
  }
  __syncthreads();

  // ---- P6: GVP2 vmu (gated) -> vA ----
  for (int o = rep; o < 16; o += 2) {
    float x = 0, y = 0, z = 0;
    const float* vh = vB + e * 99;
    for (int h = 0; h < 16; ++h) {
      float w = A.wv2[h * 16 + o];
      x += vh[h * 3 + 0] * w; y += vh[h * 3 + 1] * w; z += vh[h * 3 + 2] * w;
    }
    float g = sigmoidf_(sqrtf(x * x + y * y + z * z + 1e-8f));
    float* d = vA + e * 99 + o * 3;
    d[0] = x * g; d[1] = y * g; d[2] = z * g;
  }
  __syncthreads();

  // ---- P7: GVP2 matmul (K=128) -> relu -> tA1 (ld 128) ----
  for (int nt = 0; nt < 7; ++nt) {
    v8f acc = {};
    for (int kt = 0; kt < 4; ++kt) {
      v16bf am = load_afrag(tA2, 128, wid, kt, lane);
      v16bf bm = load_bfrag(A.pE2, nt * 4 + kt, lane);
      acc = __builtin_amdgcn_wmma_f32_16x16x32_bf16(false, am, false, bm, (short)0, acc, false, false);
    }
    int fcol  = nt * 16 + (lane & 15);
    int ebase = wid * 16 + ((lane >> 4) & 1) * 8;
    if (fcol < 100) {
      float bias = A.bs2[fcol];
#pragma unroll
      for (int r = 0; r < 8; ++r)
        tA1[(ebase + r) * 128 + fcol] = f2bf(fmaxf(acc[r] + bias, 0.f));
    }
  }
  __syncthreads();

  // ---- P8a: GVP3 vh/vn -> tA1[100..115] ----
  for (int h = rep; h < 16; h += 2) {
    float x = 0, y = 0, z = 0;
    const float* mv = vA + e * 99;
    for (int i = 0; i < 16; ++i) {
      float w = A.wh3[i * 16 + h];
      x += mv[i * 3 + 0] * w; y += mv[i * 3 + 1] * w; z += mv[i * 3 + 2] * w;
    }
    float* o = vB + e * 99 + h * 3;
    o[0] = x; o[1] = y; o[2] = z;
    tA1[e * 128 + 100 + h] = f2bf(sqrtf(x * x + y * y + z * z + 1e-8f));
  }
  __syncthreads();
  // ---- P8b: GVP3 vmu (no gate) -> vA ----
  for (int o = rep; o < 16; o += 2) {
    float x = 0, y = 0, z = 0;
    const float* vh = vB + e * 99;
    for (int h = 0; h < 16; ++h) {
      float w = A.wv3[h * 16 + o];
      x += vh[h * 3 + 0] * w; y += vh[h * 3 + 1] * w; z += vh[h * 3 + 2] * w;
    }
    float* d = vA + e * 99 + o * 3;
    d[0] = x; d[1] = y; d[2] = z;
  }
  __syncthreads();

  // ---- P9: GVP3 matmul + mean-over-edges reduction into acc_s ----
  for (int nt = 0; nt < 7; ++nt) {
    v8f acc = {};
    for (int kt = 0; kt < 4; ++kt) {
      v16bf am = load_afrag(tA1, 128, wid, kt, lane);
      v16bf bm = load_bfrag(A.pE3, nt * 4 + kt, lane);
      acc = __builtin_amdgcn_wmma_f32_16x16x32_bf16(false, am, false, bm, (short)0, acc, false, false);
    }
    int fcol = nt * 16 + (lane & 15);
    if (fcol < 100) {
      float bias = A.bs3[fcol];
      float s = 0.f;
#pragma unroll
      for (int r = 0; r < 8; ++r) s += acc[r] + bias;  // 8 edges, bias once per edge
      atomicAdd(&acc_s[fcol], s);
    }
  }
  __syncthreads();
  for (int f = tid; f < 48; f += 64) {
    float s = 0.f;
    for (int ee = 0; ee < 32; ++ee) s += vA[ee * 99 + f];
    acc_v[f] = s;
  }
  __syncthreads();

  // ---- P10: gvp_layernorm of the mean, write h_Mid ----
  if (tid == 0) {
    float vs = 0.f;
    for (int f = 0; f < 48; ++f) { float v = acc_v[f] * (1.f / 32.f); vs += v * v; }
    vs *= (1.f / 16.f);
    red[0] = 1.f / sqrtf(vs + 1e-8f);
    float mu = 0.f;
    for (int f = 0; f < 100; ++f) mu += acc_s[f] * (1.f / 32.f);
    mu *= 0.01f;
    float var = 0.f;
    for (int f = 0; f < 100; ++f) { float d = acc_s[f] * (1.f / 32.f) - mu; var += d * d; }
    var *= 0.01f;
    red[1] = mu; red[2] = 1.f / sqrtf(var + 1e-5f);
  }
  __syncthreads();
  float* outp = A.hMid + (size_t)nf * FV;
  for (int f = tid; f < 48;  f += 64) outp[f] = acc_v[f] * (1.f / 32.f) * red[0];
  for (int f = tid; f < 100; f += 64) outp[48 + f] = (acc_s[f] * (1.f / 32.f) - red[1]) * red[2];
}

// ---------------- node dense-head kernel (16 nodes / wave) ----------------
struct DhArgs {
  const float* hMid; const float* mask;
  const float *wh1, *wv1, *bs1;
  const float *wh2, *wv2, *bs2;
  const unsigned short *pD1, *pD2;
  float* outp;
};

__global__ __launch_bounds__(32) void dh_kernel(DhArgs A) {
  __shared__ unsigned short tA[16 * 160];   // dh1 A-tile: s(100)+vn(32)+pad
  __shared__ unsigned short tB[16 * 448];   // dh2 A-tile: s1(400)+vn2(32)+pad
  __shared__ float vA[16 * 96];
  __shared__ float vBv[16 * 96];
  __shared__ float sbuf[16 * 112];
  __shared__ float red[16 * 3];

  const int lane = threadIdx.x;
  const int e    = lane & 15;
  const int rep  = lane >> 4;
  const int nb   = blockIdx.x * 16;

  // P1: stage inputs
  {
    const float* x = A.hMid + (size_t)(nb + e) * FV;
    unsigned short* row = tA + e * 160;
    for (int f = rep; f < 100; f += 2) row[f] = f2bf(x[48 + f]);
    for (int f = rep; f < 28;  f += 2) row[132 + f] = 0;
    for (int f = rep; f < 16;  f += 2) tB[e * 448 + 432 + f] = 0;
    float* mv = vA + e * 96;
    for (int i = rep; i < 16; i += 2)
      for (int c = 0; c < 3; ++c) mv[i * 3 + c] = x[i * 3 + c];
  }
  __syncthreads();
  // P2: dh1 vh (16->32), vn -> tA[100..131]
  for (int h = rep; h < 32; h += 2) {
    float x = 0, y = 0, z = 0;
    const float* mv = vA + e * 96;
    for (int i = 0; i < 16; ++i) {
      float w = A.wh1[i * 32 + h];
      x += mv[i * 3 + 0] * w; y += mv[i * 3 + 1] * w; z += mv[i * 3 + 2] * w;
    }
    float* o = vBv + e * 96 + h * 3;
    o[0] = x; o[1] = y; o[2] = z;
    tA[e * 160 + 100 + h] = f2bf(sqrtf(x * x + y * y + z * z + 1e-8f));
  }
  __syncthreads();
  // P3: dh1 vmu (gated, 32 out) -> vA
  for (int o = rep; o < 32; o += 2) {
    float x = 0, y = 0, z = 0;
    const float* vh = vBv + e * 96;
    for (int h = 0; h < 32; ++h) {
      float w = A.wv1[h * 32 + o];
      x += vh[h * 3 + 0] * w; y += vh[h * 3 + 1] * w; z += vh[h * 3 + 2] * w;
    }
    float g = sigmoidf_(sqrtf(x * x + y * y + z * z + 1e-8f));
    float* d = vA + e * 96 + o * 3;
    d[0] = x * g; d[1] = y * g; d[2] = z * g;
  }
  __syncthreads();
  // P4: dh1 matmul (K=160, N=400) -> relu -> tB
  for (int nt = 0; nt < 25; ++nt) {
    v8f acc = {};
    for (int kt = 0; kt < 5; ++kt) {
      v16bf am = load_afrag(tA, 160, 0, kt, lane);
      v16bf bm = load_bfrag(A.pD1, nt * 5 + kt, lane);
      acc = __builtin_amdgcn_wmma_f32_16x16x32_bf16(false, am, false, bm, (short)0, acc, false, false);
    }
    int fcol  = nt * 16 + (lane & 15);
    int ebase = ((lane >> 4) & 1) * 8;
    float bias = A.bs1[fcol];
#pragma unroll
    for (int r = 0; r < 8; ++r)
      tB[(ebase + r) * 448 + fcol] = f2bf(fmaxf(acc[r] + bias, 0.f));
  }
  __syncthreads();
  // P5: dh2 vh (32->32), vn2 -> tB[400..431]
  for (int h = rep; h < 32; h += 2) {
    float x = 0, y = 0, z = 0;
    const float* mv = vA + e * 96;
    for (int i = 0; i < 32; ++i) {
      float w = A.wh2[i * 32 + h];
      x += mv[i * 3 + 0] * w; y += mv[i * 3 + 1] * w; z += mv[i * 3 + 2] * w;
    }
    float* o = vBv + e * 96 + h * 3;
    o[0] = x; o[1] = y; o[2] = z;
    tB[e * 448 + 400 + h] = f2bf(sqrtf(x * x + y * y + z * z + 1e-8f));
  }
  __syncthreads();
  // P6: dh2 vmu (16 out, no gate) -> vA
  for (int o = rep; o < 16; o += 2) {
    float x = 0, y = 0, z = 0;
    const float* vh = vBv + e * 96;
    for (int h = 0; h < 32; ++h) {
      float w = A.wv2[h * 16 + o];
      x += vh[h * 3 + 0] * w; y += vh[h * 3 + 1] * w; z += vh[h * 3 + 2] * w;
    }
    float* d = vA + e * 96 + o * 3;
    d[0] = x; d[1] = y; d[2] = z;
  }
  __syncthreads();
  // P7: dh2 matmul (K=448) -> sbuf
  for (int nt = 0; nt < 7; ++nt) {
    v8f acc = {};
    for (int kt = 0; kt < 14; ++kt) {
      v16bf am = load_afrag(tB, 448, 0, kt, lane);
      v16bf bm = load_bfrag(A.pD2, nt * 14 + kt, lane);
      acc = __builtin_amdgcn_wmma_f32_16x16x32_bf16(false, am, false, bm, (short)0, acc, false, false);
    }
    int fcol  = nt * 16 + (lane & 15);
    int ebase = ((lane >> 4) & 1) * 8;
    if (fcol < 100) {
      float bias = A.bs2[fcol];
#pragma unroll
      for (int r = 0; r < 8; ++r) sbuf[(ebase + r) * 112 + fcol] = acc[r] + bias;
    }
  }
  __syncthreads();
  // P8: gvp_layernorm + mask, write h_V
  if (lane < 16) {
    const float* v = vA + lane * 96;
    float vs = 0.f;
    for (int f = 0; f < 48; ++f) vs += v[f] * v[f];
    vs *= (1.f / 16.f);
    red[lane * 3 + 0] = 1.f / sqrtf(vs + 1e-8f);
    const float* s = sbuf + lane * 112;
    float mu = 0.f;
    for (int f = 0; f < 100; ++f) mu += s[f];
    mu *= 0.01f;
    float var = 0.f;
    for (int f = 0; f < 100; ++f) { float d = s[f] - mu; var += d * d; }
    var *= 0.01f;
    red[lane * 3 + 1] = mu; red[lane * 3 + 2] = 1.f / sqrtf(var + 1e-5f);
  }
  __syncthreads();
  {
    const float m = A.mask[nb + e];
    float* o = A.outp + (size_t)(nb + e) * FV;
    const float* v = vA + e * 96;
    const float* s = sbuf + e * 112;
    for (int f = rep; f < 48;  f += 2) o[f] = m * v[f] * red[e * 3 + 0];
    for (int f = rep; f < 100; f += 2) o[48 + f] = m * (s[f] - red[e * 3 + 1]) * red[e * 3 + 2];
  }
}

// ---------------- host launch ----------------
extern "C" void kernel_launch(void* const* d_in, const int* in_sizes, int n_in,
                              void* d_out, int out_size, void* d_ws, size_t ws_size,
                              hipStream_t stream) {
  (void)in_sizes; (void)n_in; (void)out_size; (void)ws_size;
  const float* hV0 = (const float*)d_in[0];
  const float* hS  = (const float*)d_in[1];
  const float* hE  = (const float*)d_in[2];
  const int*   Eix = (const int*)d_in[3];
  const float* msk = (const float*)d_in[4];
  // params flattened (jax tree order): per layer: W_EV[0..2]{bs,wh,ws,wv}, W_dh[0..1]{bs,wh,ws,wv}
  auto W = [&](int l, int grp, int g, int which) -> const float* {
    return (const float*)d_in[5 + l * 20 + (grp ? 12 : 0) + g * 4 + which];
  }; // which: 0=bs, 1=wh, 2=ws, 3=wv

  const size_t BN = (size_t)BB * NN;
  auto al = [](size_t x) { return (x + 255) & ~(size_t)255; };
  char* ws = (char*)d_ws;
  size_t off = 0;
  float* hVcur = (float*)(ws + off); off += al(BN * FV * 4);
  float* hMid  = (float*)(ws + off); off += al(BN * FV * 4);
  unsigned short *pE1[3], *pE2[3], *pE3[3], *pD1[3], *pD2[3];
  for (int l = 0; l < 3; ++l) {
    pE1[l] = (unsigned short*)(ws + off); off += al((size_t)13 * 7 * 512 * 2);
    pE2[l] = (unsigned short*)(ws + off); off += al((size_t)4 * 7 * 512 * 2);
    pE3[l] = (unsigned short*)(ws + off); off += al((size_t)4 * 7 * 512 * 2);
    pD1[l] = (unsigned short*)(ws + off); off += al((size_t)5 * 25 * 512 * 2);
    pD2[l] = (unsigned short*)(ws + off); off += al((size_t)14 * 7 * 512 * 2);
  }

  hipMemcpyAsync(hVcur, hV0, BN * FV * 4, hipMemcpyDeviceToDevice, stream);

  for (int l = 0; l < 3; ++l) {
    pack_kernel<<<13 * 7, 512, 0, stream>>>(W(l, 0, 0, 2), 393, 100, 13, pE1[l]);
    pack_kernel<<<4 * 7,  512, 0, stream>>>(W(l, 0, 1, 2), 116, 100, 4,  pE2[l]);
    pack_kernel<<<4 * 7,  512, 0, stream>>>(W(l, 0, 2, 2), 116, 100, 4,  pE3[l]);
    pack_kernel<<<5 * 25, 512, 0, stream>>>(W(l, 1, 0, 2), 132, 400, 5,  pD1[l]);
    pack_kernel<<<14 * 7, 512, 0, stream>>>(W(l, 1, 1, 2), 432, 100, 14, pD2[l]);
  }

  for (int l = 0; l < 3; ++l) {
    EdgeArgs ea{ hVcur, hV0, hS, hE, Eix, msk,
                 W(l, 0, 0, 1), W(l, 0, 0, 3), W(l, 0, 0, 0),
                 W(l, 0, 1, 1), W(l, 0, 1, 3), W(l, 0, 1, 0),
                 W(l, 0, 2, 1), W(l, 0, 2, 3), W(l, 0, 2, 0),
                 pE1[l], pE2[l], pE3[l], hMid };
    edge_kernel<<<(int)BN, 64, 0, stream>>>(ea);

    DhArgs da{ hMid, msk,
               W(l, 1, 0, 1), W(l, 1, 0, 3), W(l, 1, 0, 0),
               W(l, 1, 1, 1), W(l, 1, 1, 3), W(l, 1, 1, 0),
               pD1[l], pD2[l],
               (l == 2) ? (float*)d_out : hVcur };
    dh_kernel<<<(int)(BN / 16), 32, 0, stream>>>(da);
  }
}
